// VQVAE_52699248722363
// MI455X (gfx1250) — matrix operand order repaired and checked by reading, not compiled, not run on hardware
//
#include <hip/hip_runtime.h>
#include <hip/hip_bf16.h>
#include <math.h>

typedef __attribute__((ext_vector_type(16))) __bf16 v16bf;
typedef __attribute__((ext_vector_type(8)))  float  v8f;
typedef __attribute__((ext_vector_type(4)))  unsigned int v4u;
typedef __attribute__((ext_vector_type(8)))  int v8i;
typedef __attribute__((ext_vector_type(4)))  int v4i;

#if defined(__has_builtin)
#if __has_builtin(__builtin_amdgcn_tensor_load_to_lds) && \
    __has_builtin(__builtin_amdgcn_s_wait_tensorcnt)
#define HAVE_TDM 1
#endif
#endif
#ifndef HAVE_TDM
#define HAVE_TDM 0
#endif

union BF16Frag { v16bf v; uint4 q[2]; unsigned short u[16]; };
union US8 { unsigned short s[8]; uint4 q; };

__device__ __forceinline__ unsigned short f2bf(float f) {
  unsigned u = __float_as_uint(f);
  return (unsigned short)((u + 0x7fffu + ((u >> 16) & 1u)) >> 16);   // RNE
}
__device__ __forceinline__ float bf2f(unsigned short h) {
  return __uint_as_float(((unsigned)h) << 16);
}

#if HAVE_TDM
// TDM 2D tile load: [rows x 32] bf16 tile of a [.. x strideElems] matrix into
// LDS with 16B row padding (64B data + 16B pad = 80B = 40 ushort rows).
// D# per CDNA5 ISA ch.8: group0 {count, lds_addr, global_addr, type=2},
// group1 {data_size=2B, pad_enable, pad_interval=3(16DW), pad_amount=3(4DW),
//         tensor_dim0/1 (OOB zero-fill), tile_dim0=32, tile_dim1=rows,
//         tensor_dim0_stride}.
// amdgpu-toolchain (clang-23) 6-arg builtin form.
__device__ __forceinline__ void tdm_load_tile_b16(
    unsigned ldsAddr, const void* gaddr, unsigned remCols, unsigned remRows,
    unsigned rows, unsigned strideElems)
{
  unsigned long long ga = (unsigned long long)(size_t)gaddr;
  v4u g0; v8i g1;
  v4i g2 = {0, 0, 0, 0}, g3 = {0, 0, 0, 0};
  v8i g4 = {0, 0, 0, 0, 0, 0, 0, 0};
  g0[0] = 1u;                                           // count=1, user mode
  g0[1] = ldsAddr;                                      // lds_addr[31:0]
  g0[2] = (unsigned)(ga & 0xffffffffu);                 // global_addr[31:0]
  g0[3] = (unsigned)((ga >> 32) & 0x01ffffffu) | (2u << 30);  // [56:32]|type=2
  g1[0] = (int)((1u << 16) | (1u << 20) | (3u << 22) | (3u << 25));
  g1[1] = (int)((remCols & 0xffffu) << 16);             // tensor_dim0 lo16
  g1[2] = (int)(((remCols >> 16) & 0xffffu) | ((remRows & 0xffffu) << 16));
  g1[3] = (int)(((remRows >> 16) & 0xffffu) | (32u << 16));   // tile_dim0=32
  g1[4] = (int)(rows & 0xffffu);                        // tile_dim1, tile_dim2=0
  g1[5] = (int)strideElems;                             // dim0_stride lo32
  g1[6] = 0;
  g1[7] = 0;
  __builtin_amdgcn_tensor_load_to_lds(g0, g1, g2, g3, g4, 0);
}
#endif

// ---------------------------------------------------------------------------
// Implicit-GEMM conv (MODE 0) / ConvTranspose2d s2 k4 p1 (MODE 1).
//   M = B*Hout*Wout (multiple of 64), N = Cout, K = Cin*KH*KW.
//   Block 256 thr = 8 waves; tile 64(M) x 64(N); K step 32.
//   Wave (4x2) owns 16(M) x 32(N): one A fragment feeds two WMMAs.
//   B tile staged by the Tensor Data Mover (TENSORcnt) when available.
// ---------------------------------------------------------------------------
#define TM 64
#define TN 64
#define TK 32
#define ASTR 40
#define BSTR 40

template <int KH, int KW, int MODE>
__global__ __launch_bounds__(256) void conv_gemm_bf16(
    const unsigned short* __restrict__ in, const unsigned short* __restrict__ wBT,
    const float* __restrict__ bias,
    unsigned short* __restrict__ out_bf, float* __restrict__ out_f32,
    const unsigned short* __restrict__ resid,
    int Cin, int Hin, int Win, int Cout,
    int stride, int pad, int wshift, int hwshift, int relu_in, int act)
{
  constexpr int KHW = KH * KW;
  __shared__ __align__(16) unsigned short As[TM][ASTR];
  __shared__ __align__(16) unsigned short Bs[TN][BSTR];

  const int Ktot = Cin * KHW;
  const int m0   = blockIdx.x * TM;
  const int n0   = blockIdx.y * TN;
  const int tid  = threadIdx.x;
  const int wave = tid >> 5;
  const int lane = tid & 31;
  const int hf   = lane >> 4;
  const int l15  = lane & 15;
  const int wr   = (wave & 3) * 16;
  const int wc   = (wave >> 2) * 32;
  const int Wout = 1 << wshift;

  // ---- per-thread A-tile coordinates (hoisted out of the K loop) ----
  const int arow = tid >> 2;          // one LDS row per 4 threads
  const int kl0  = (tid & 3) * 8;     // 8 consecutive k per thread
  const int am   = m0 + arow;
  const int ab   = am >> hwshift;
  const int arem = am & ((1 << hwshift) - 1);
  const int aoy  = arem >> wshift;
  const int aox  = arem & (Wout - 1);

  v8f acc0 = {}, acc1 = {};

  for (int k0 = 0; k0 < Ktot; k0 += TK) {
    if (tid == 0 && (k0 + TK) < Ktot)   // prefetch next weight tile (gfx1250)
      __builtin_prefetch(wBT + (size_t)n0 * Ktot + k0 + TK, 0, 1);

    // ---- A tile: im2col gather, one b128 LDS store per thread ----
    US8 av;
#pragma unroll
    for (int j = 0; j < 8; ++j) {
      int k = k0 + kl0 + j;
      unsigned short val = 0;
      if (k < Ktot) {
        int ci = k / KHW;               // compile-time divisor
        int r  = k - ci * KHW;
        int ky = r / KW, kx = r - ky * KW;
        int iy, ix; bool ok;
        if (MODE == 0) {
          iy = aoy * stride + ky - pad;
          ix = aox * stride + kx - pad;
          ok = (iy >= 0 && iy < Hin && ix >= 0 && ix < Win);
        } else {                         // lhs_dilation=2, pad=2
          int ty = aoy + ky - 2, tx = aox + kx - 2;
          ok = (ty >= 0 && tx >= 0 && !(ty & 1) && !(tx & 1));
          iy = ty >> 1; ix = tx >> 1;
          ok = ok && (iy < Hin) && (ix < Win);
        }
        if (ok) {
          val = in[((ab * Cin + ci) * Hin + iy) * Win + ix];
          if (relu_in && (val & 0x8000u)) val = 0;
        }
      }
      av.s[j] = val;
    }
    *(uint4*)&As[arow][kl0] = av.q;

#if HAVE_TDM
    // ---- B tile: Tensor Data Mover DMA, wave 0 issues + waits TENSORcnt ----
    if (wave == 0) {
      tdm_load_tile_b16((unsigned)(size_t)&Bs[0][0],
                        wBT + (size_t)n0 * Ktot + k0,
                        (unsigned)(Ktot - k0), (unsigned)(Cout - n0),
                        TN, (unsigned)Ktot);
      __builtin_amdgcn_s_wait_tensorcnt(0);
    }
#else
    // ---- B tile: contiguous b128 global load -> b128 LDS store ----
    {
      const int brow = tid >> 2;
      const int bn   = n0 + brow;
      uint4 bv = make_uint4(0u, 0u, 0u, 0u);
      if (bn < Cout && (k0 + kl0) < Ktot)  // Ktot % 8 == 0 for all layers
        bv = *(const uint4*)(wBT + (size_t)bn * Ktot + k0 + kl0);
      *(uint4*)&Bs[brow][kl0] = bv;
    }
#endif
    __syncthreads();

    // ---- fragments: pure b128 LDS loads ----
    BF16Frag fa, fb0, fb1;
    fa.q[0]  = *(const uint4*)&As[wr + l15][hf * 8];
    fa.q[1]  = *(const uint4*)&As[wr + l15][16 + hf * 8];
    fb0.q[0] = *(const uint4*)&Bs[wc + l15][hf * 8];
    fb0.q[1] = *(const uint4*)&Bs[wc + l15][16 + hf * 8];
    fb1.q[0] = *(const uint4*)&Bs[wc + 16 + l15][hf * 8];
    fb1.q[1] = *(const uint4*)&Bs[wc + 16 + l15][16 + hf * 8];

    acc0 = __builtin_amdgcn_wmma_f32_16x16x32_bf16(false, fa.v, false, fb0.v,
                                                   (short)0, acc0, false, false);
    acc1 = __builtin_amdgcn_wmma_f32_16x16x32_bf16(false, fa.v, false, fb1.v,
                                                   (short)0, acc1, false, false);
    __syncthreads();
  }

  // ---- epilogue: bias + residual + activation ----
#pragma unroll
  for (int t = 0; t < 2; ++t) {
    v8f& acc = t ? acc1 : acc0;
    int n = n0 + wc + t * 16 + l15;
    if (n >= Cout) continue;
    float bv = bias[n];
#pragma unroll
    for (int r = 0; r < 8; ++r) {
      int m    = m0 + wr + r + 8 * hf;
      int b    = m >> hwshift;
      int rem  = m & ((1 << hwshift) - 1);
      int oidx = ((b * Cout + n) << hwshift) + rem;   // same spatial layout
      float v = acc[r] + bv;
      if (resid) v += bf2f(resid[oidx]);
      if (act == 1) v = fmaxf(v, 0.0f);
      else if (act == 2) v = tanhf(v);
      if (out_bf)  out_bf[oidx]  = f2bf(v);
      if (out_f32) out_f32[oidx] = v;
    }
  }
}

// ---------------------------------------------------------------------------
// VQ argmin_j( ||e_j||^2 - 2 z.e_j ) : [131072,64] x [64,512] via WMMA.
// embN is bf16 row-major [512][64] -> B fragments are two b128 global loads.
// ---------------------------------------------------------------------------
__global__ __launch_bounds__(256) void vq_argmin_wmma(
    const unsigned short* __restrict__ zbf,   // [32,64,64,64] NCHW bf16
    const unsigned short* __restrict__ embN,  // [512][64] bf16
    const float* __restrict__ enorm,          // [512]
    int* __restrict__ idx_out, float* __restrict__ idx_f_out)
{
  __shared__ __align__(16) unsigned short As[128][72];
  const int row0 = blockIdx.x * 128;
  const int tid  = threadIdx.x;
  const int wave = tid >> 5, lane = tid & 31;
  const int hf   = lane >> 4, l15 = lane & 15;
  const int wr   = wave * 16;

  // zf[n][c] = z[b][c][h][w]; iterate n-fastest for coalesced global reads
  for (int i = tid; i < 128 * 64; i += 256) {
    int c = i >> 7, ml = i & 127;
    int n = row0 + ml;
    int b = n >> 12, rem = n & 4095;
    As[ml][c] = zbf[((b * 64 + c) << 12) + rem];
  }
  __syncthreads();

  BF16Frag fa0, fa1;
  fa0.q[0] = *(const uint4*)&As[wr + l15][hf * 8];
  fa0.q[1] = *(const uint4*)&As[wr + l15][16 + hf * 8];
  fa1.q[0] = *(const uint4*)&As[wr + l15][32 + hf * 8];
  fa1.q[1] = *(const uint4*)&As[wr + l15][48 + hf * 8];

  float minv[8]; int mini[8];
#pragma unroll
  for (int r = 0; r < 8; ++r) { minv[r] = 3.4e38f; mini[r] = 0; }

  for (int ct = 0; ct < 32; ++ct) {
    const int ncol = ct * 16 + l15;
    BF16Frag fb0, fb1;
    fb0.q[0] = *(const uint4*)(embN + ncol * 64 + hf * 8);
    fb0.q[1] = *(const uint4*)(embN + ncol * 64 + 16 + hf * 8);
    fb1.q[0] = *(const uint4*)(embN + ncol * 64 + 32 + hf * 8);
    fb1.q[1] = *(const uint4*)(embN + ncol * 64 + 48 + hf * 8);
    v8f acc = {};
    acc = __builtin_amdgcn_wmma_f32_16x16x32_bf16(false, fa0.v, false, fb0.v,
                                                  (short)0, acc, false, false);
    acc = __builtin_amdgcn_wmma_f32_16x16x32_bf16(false, fa1.v, false, fb1.v,
                                                  (short)0, acc, false, false);
    float en = enorm[ncol];
#pragma unroll
    for (int r = 0; r < 8; ++r) {
      float d = en - 2.0f * acc[r];
      if (d < minv[r]) { minv[r] = d; mini[r] = ncol; }
    }
  }
  // min-reduce over the 16 lanes holding the same rows (first-index ties)
#pragma unroll
  for (int off = 1; off < 16; off <<= 1) {
#pragma unroll
    for (int r = 0; r < 8; ++r) {
      float ov = __shfl_xor(minv[r], off, 32);
      int   oi = __shfl_xor(mini[r], off, 32);
      if (ov < minv[r] || (ov == minv[r] && oi < mini[r])) { minv[r] = ov; mini[r] = oi; }
    }
  }
  if (l15 == 0) {
#pragma unroll
    for (int r = 0; r < 8; ++r) {
      int m = row0 + wr + r + 8 * hf;
      idx_out[m]   = mini[r];
      idx_f_out[m] = (float)mini[r];
    }
  }
}

// z_q gather (faithful flat-order reinterpretation), z_st = z_q, loss accum
__global__ __launch_bounds__(256) void vq_zq_loss(
    const float* __restrict__ zf32, const float* __restrict__ emb,
    const int* __restrict__ idx, unsigned short* __restrict__ zst_bf,
    float* __restrict__ lacc)
{
  __shared__ float red[256];
  int f = blockIdx.x * 256 + threadIdx.x;           // 0 .. 8388607
  int b = f >> 18, rem = f & 262143;
  int c = rem >> 12, rem2 = rem & 4095;
  int h = rem2 >> 6, w = rem2 & 63;
  int n = (b << 12) + (c << 6) + h;                 // faithful .view() bug
  float zq = emb[idx[n] * 64 + w];
  float d  = zq - zf32[f];
  zst_bf[f] = f2bf(zq);                             // z_st value == z_q
  red[threadIdx.x] = d * d;
  __syncthreads();
  for (int s = 128; s > 0; s >>= 1) {
    if (threadIdx.x < s) red[threadIdx.x] += red[threadIdx.x + s];
    __syncthreads();
  }
  if (threadIdx.x == 0) atomicAdd(lacc, red[0]);
}

// ------------------------------ prep kernels -------------------------------
__global__ void k_f32_to_bf16(const float* __restrict__ s,
                              unsigned short* __restrict__ d, int n) {
  int i = blockIdx.x * 256 + threadIdx.x;
  if (i < n) d[i] = f2bf(s[i]);
}
// ConvT weights -> [Cout][K], K = ci*16 + ky*4 + kx, spatially flipped
__global__ void k_prep_w_convT(const float* __restrict__ w,
                               unsigned short* __restrict__ wBT,
                               int Cin, int Cout) {
  int i = blockIdx.x * 256 + threadIdx.x;
  if (i >= Cin * 16 * Cout) return;
  int Ktot = Cin * 16;
  int co = i / Ktot, k = i - co * Ktot;
  int ci = k >> 4, r = k & 15, ky = r >> 2, kx = r & 3;
  wBT[i] = f2bf(w[((ci * Cout + co) * 4 + (3 - ky)) * 4 + (3 - kx)]);
}
__global__ void k_enorm(const float* __restrict__ emb, float* __restrict__ enorm) {
  int j = threadIdx.x;                                   // 512 codes
  float s = 0.0f;
  for (int c = 0; c < 64; ++c) { float v = emb[j * 64 + c]; s += v * v; }
  enorm[j] = s;
}
__global__ void k_zero_f32(float* p) { if (threadIdx.x == 0 && blockIdx.x == 0) *p = 0.0f; }
__global__ void k_finalize_loss(const float* __restrict__ lacc, float* __restrict__ out) {
  if (threadIdx.x == 0 && blockIdx.x == 0)
    *out = 1.25f * (*lacc) / 8388608.0f;                 // (1 + 0.25) * mean
}

// ---------------------------------------------------------------------------
extern "C" void kernel_launch(void* const* d_in, const int* in_sizes, int n_in,
                              void* d_out, int out_size, void* d_ws, size_t ws_size,
                              hipStream_t stream) {
  const float* x     = (const float*)d_in[0];
  const float* ew1   = (const float*)d_in[1];
  const float* eb1   = (const float*)d_in[2];
  const float* ew2   = (const float*)d_in[3];
  const float* eb2   = (const float*)d_in[4];
  const float* er1w3 = (const float*)d_in[5];
  const float* er1b3 = (const float*)d_in[6];
  const float* er1w1 = (const float*)d_in[7];
  const float* er1b1 = (const float*)d_in[8];
  const float* er2w3 = (const float*)d_in[9];
  const float* er2b3 = (const float*)d_in[10];
  const float* er2w1 = (const float*)d_in[11];
  const float* er2b1 = (const float*)d_in[12];
  const float* eow   = (const float*)d_in[13];
  const float* eob   = (const float*)d_in[14];
  const float* emb   = (const float*)d_in[15];
  const float* dw0   = (const float*)d_in[16];
  const float* db0   = (const float*)d_in[17];
  const float* dr1w3 = (const float*)d_in[18];
  const float* dr1b3 = (const float*)d_in[19];
  const float* dr1w1 = (const float*)d_in[20];
  const float* dr1b1 = (const float*)d_in[21];
  const float* dr2w3 = (const float*)d_in[22];
  const float* dr2b3 = (const float*)d_in[23];
  const float* dr2w1 = (const float*)d_in[24];
  const float* dr2b1 = (const float*)d_in[25];
  const float* dt1w  = (const float*)d_in[26];
  const float* dt1b  = (const float*)d_in[27];
  const float* dt2w  = (const float*)d_in[28];
  const float* dt2b  = (const float*)d_in[29];

  float* out_x    = (float*)d_out;                 // [32,3,256,256]
  float* out_loss = (float*)d_out + 6291456;       // [1]
  float* out_idx  = (float*)d_out + 6291457;       // [131072]

  size_t cur = 0;
  auto alloc = [&](size_t bytes) -> void* {
    void* p = (char*)d_ws + cur;
    cur += (bytes + 255) & ~(size_t)255;
    return p;
  };
  unsigned short* xbf  = (unsigned short*)alloc(6291456ull * 2);
  unsigned short* act1 = (unsigned short*)alloc(67108864ull * 2);  // 128x128 stage
  unsigned short* act2 = (unsigned short*)alloc(16777216ull * 2);  // 64x64 stage
  unsigned short* act3 = (unsigned short*)alloc(16777216ull * 2);
  unsigned short* act4 = (unsigned short*)alloc(16777216ull * 2);
  unsigned short* zbf  = (unsigned short*)alloc(8388608ull * 2);
  float*          zf32 = (float*)alloc(8388608ull * 4);
  unsigned short* zst  = (unsigned short*)alloc(8388608ull * 2);
  unsigned short* embN = (unsigned short*)alloc(512ull * 64 * 2);
  float*          enrm = (float*)alloc(512 * 4);
  int*            idx  = (int*)alloc(131072 * 4);
  float*          lacc = (float*)alloc(256);
  unsigned short* wb_ew1   = (unsigned short*)alloc(48ull   * 128 * 2);
  unsigned short* wb_ew2   = (unsigned short*)alloc(2048ull * 128 * 2);
  unsigned short* wb_er1w3 = (unsigned short*)alloc(1152ull * 128 * 2);
  unsigned short* wb_er1w1 = (unsigned short*)alloc(128ull  * 128 * 2);
  unsigned short* wb_er2w3 = (unsigned short*)alloc(1152ull * 128 * 2);
  unsigned short* wb_er2w1 = (unsigned short*)alloc(128ull  * 128 * 2);
  unsigned short* wb_eow   = (unsigned short*)alloc(128ull  * 64  * 2);
  unsigned short* wb_dw0   = (unsigned short*)alloc(576ull  * 128 * 2);
  unsigned short* wb_dr1w3 = (unsigned short*)alloc(1152ull * 128 * 2);
  unsigned short* wb_dr1w1 = (unsigned short*)alloc(128ull  * 128 * 2);
  unsigned short* wb_dr2w3 = (unsigned short*)alloc(1152ull * 128 * 2);
  unsigned short* wb_dr2w1 = (unsigned short*)alloc(128ull  * 128 * 2);
  unsigned short* wb_dt1   = (unsigned short*)alloc(2048ull * 128 * 2);
  unsigned short* wb_dt2   = (unsigned short*)alloc(2048ull * 3   * 2);

  auto gb = [](int n) { return (n + 255) / 256; };
  // conv weights are already [Cout][Cin*KH*KW] row-major == [Cout][K] B^T
  k_f32_to_bf16<<<gb(6291456), 256, 0, stream>>>(x, xbf, 6291456);
  k_f32_to_bf16<<<gb(48 * 128), 256, 0, stream>>>(ew1, wb_ew1, 48 * 128);
  k_f32_to_bf16<<<gb(2048 * 128), 256, 0, stream>>>(ew2, wb_ew2, 2048 * 128);
  k_f32_to_bf16<<<gb(1152 * 128), 256, 0, stream>>>(er1w3, wb_er1w3, 1152 * 128);
  k_f32_to_bf16<<<gb(128 * 128), 256, 0, stream>>>(er1w1, wb_er1w1, 128 * 128);
  k_f32_to_bf16<<<gb(1152 * 128), 256, 0, stream>>>(er2w3, wb_er2w3, 1152 * 128);
  k_f32_to_bf16<<<gb(128 * 128), 256, 0, stream>>>(er2w1, wb_er2w1, 128 * 128);
  k_f32_to_bf16<<<gb(128 * 64), 256, 0, stream>>>(eow, wb_eow, 128 * 64);
  k_f32_to_bf16<<<gb(576 * 128), 256, 0, stream>>>(dw0, wb_dw0, 576 * 128);
  k_f32_to_bf16<<<gb(1152 * 128), 256, 0, stream>>>(dr1w3, wb_dr1w3, 1152 * 128);
  k_f32_to_bf16<<<gb(128 * 128), 256, 0, stream>>>(dr1w1, wb_dr1w1, 128 * 128);
  k_f32_to_bf16<<<gb(1152 * 128), 256, 0, stream>>>(dr2w3, wb_dr2w3, 1152 * 128);
  k_f32_to_bf16<<<gb(128 * 128), 256, 0, stream>>>(dr2w1, wb_dr2w1, 128 * 128);
  k_f32_to_bf16<<<gb(512 * 64), 256, 0, stream>>>(emb, embN, 512 * 64);
  k_prep_w_convT<<<gb(2048 * 128), 256, 0, stream>>>(dt1w, wb_dt1, 128, 128);
  k_prep_w_convT<<<gb(2048 * 3), 256, 0, stream>>>(dt2w, wb_dt2, 128, 3);
  k_enorm<<<1, 512, 0, stream>>>(emb, enrm);
  k_zero_f32<<<1, 64, 0, stream>>>(lacc);

  auto conv = [&](const unsigned short* in, const unsigned short* wBT, const float* bias,
                  unsigned short* obf, float* of32, const unsigned short* resid,
                  int Cin, int Hin, int Win, int Cout, int Hout,
                  int KH, int s, int p, int mode, int wsh, int hwsh,
                  int relu_in, int act) {
    int M = 32 * (1 << hwsh);
    dim3 g(M / TM, (Cout + TN - 1) / TN);
    if (mode == 1)
      conv_gemm_bf16<4, 4, 1><<<g, 256, 0, stream>>>(in, wBT, bias, obf, of32, resid,
          Cin, Hin, Win, Cout, s, p, wsh, hwsh, relu_in, act);
    else if (KH == 4)
      conv_gemm_bf16<4, 4, 0><<<g, 256, 0, stream>>>(in, wBT, bias, obf, of32, resid,
          Cin, Hin, Win, Cout, s, p, wsh, hwsh, relu_in, act);
    else if (KH == 3)
      conv_gemm_bf16<3, 3, 0><<<g, 256, 0, stream>>>(in, wBT, bias, obf, of32, resid,
          Cin, Hin, Win, Cout, s, p, wsh, hwsh, relu_in, act);
    else
      conv_gemm_bf16<1, 1, 0><<<g, 256, 0, stream>>>(in, wBT, bias, obf, of32, resid,
          Cin, Hin, Win, Cout, s, p, wsh, hwsh, relu_in, act);
  };

  // ---------------- encoder ----------------
  conv(xbf,  wb_ew1,   eb1,   act1, nullptr, nullptr,   3, 256, 256, 128, 128, 4, 2, 1, 0, 7, 14, 0, 1);
  conv(act1, wb_ew2,   eb2,   act2, nullptr, nullptr, 128, 128, 128, 128,  64, 4, 2, 1, 0, 6, 12, 0, 1);
  conv(act2, wb_er1w3, er1b3, act3, nullptr, nullptr, 128,  64,  64, 128,  64, 3, 1, 1, 0, 6, 12, 1, 0);
  conv(act3, wb_er1w1, er1b1, act4, nullptr, act2,    128,  64,  64, 128,  64, 1, 1, 0, 0, 6, 12, 1, 0);
  conv(act4, wb_er2w3, er2b3, act3, nullptr, nullptr, 128,  64,  64, 128,  64, 3, 1, 1, 0, 6, 12, 1, 0);
  conv(act3, wb_er2w1, er2b1, act2, nullptr, act4,    128,  64,  64, 128,  64, 1, 1, 0, 0, 6, 12, 1, 0);
  conv(act2, wb_eow,   eob,   zbf,  zf32,    nullptr, 128,  64,  64,  64,  64, 1, 1, 0, 0, 6, 12, 0, 0);

  // ---------------- vector quantization ----------------
  vq_argmin_wmma<<<1024, 256, 0, stream>>>(zbf, embN, enrm, idx, out_idx);
  vq_zq_loss<<<32768, 256, 0, stream>>>(zf32, emb, idx, zst, lacc);
  k_finalize_loss<<<1, 64, 0, stream>>>(lacc, out_loss);

  // ---------------- decoder ----------------
  conv(zst,  wb_dw0,   db0,   act3, nullptr, nullptr,  64,  64,  64, 128,  64, 3, 1, 1, 0, 6, 12, 0, 0);
  conv(act3, wb_dr1w3, dr1b3, act4, nullptr, nullptr, 128,  64,  64, 128,  64, 3, 1, 1, 0, 6, 12, 1, 0);
  conv(act4, wb_dr1w1, dr1b1, act2, nullptr, act3,    128,  64,  64, 128,  64, 1, 1, 0, 0, 6, 12, 1, 0);
  conv(act2, wb_dr2w3, dr2b3, act4, nullptr, nullptr, 128,  64,  64, 128,  64, 3, 1, 1, 0, 6, 12, 1, 0);
  conv(act4, wb_dr2w1, dr2b1, act3, nullptr, act2,    128,  64,  64, 128,  64, 1, 1, 0, 0, 6, 12, 1, 0);
  conv(act3, wb_dt1,   dt1b,  act1, nullptr, nullptr, 128,  64,  64, 128, 128, 4, 2, 1, 1, 7, 14, 0, 1);
  conv(act1, wb_dt2,   dt2b,  nullptr, out_x, nullptr,128, 128, 128,   3, 256, 4, 2, 1, 1, 8, 16, 0, 2);
}
